// WeightedReadout_50878182588792
// MI455X (gfx1250) — matrix operand order repaired and checked by reading, not compiled, not run on hardware
//
#include <hip/hip_runtime.h>
#include <hip/hip_bf16.h>

// ---------------------------------------------------------------------------
// WeightedReadout fused pipeline for MI455X (gfx1250, wave32, WMMA f16)
//   N=800000 nodes, D=128, G=4096 segments (sorted batch_index)
//
//   k_init    : zero out / segment scratch
//   k_pack    : f32 weights -> f16, pre-swizzled into WMMA B-fragment layout
//   k_logits  : w = silu(X@Ww1+bw1)@Ww2+bw2 (WMMA) + segment atomicMax
//   k_expsum  : w_exp = exp(w - max), segment sums
//   k_pool    : pf = silu(X@Wf1+bf1)@Wf2+bf2 recomputed (never materialized:
//               saves a 410MB HBM write), sorted-run-compressed weighted
//               segment sum into out[G,D].
//
//   Weights are staged into LDS once per workgroup with CDNA5
//   global_load_async_to_lds_b128 (ASYNCcnt), overlapped with the f32->f16
//   X-tile conversion; B fragments then come from LDS (ds_load_b128).
// ---------------------------------------------------------------------------

#define NN      800000
#define DD      128
#define GG      4096
#define TILES   (NN / 128)          // 6250, N divisible by 128
#define ROWH    136                 // padded LDS row stride in halves (128+8)

typedef _Float16 v16h __attribute__((ext_vector_type(16)));
typedef _Float16 v8h  __attribute__((ext_vector_type(8)));
typedef _Float16 v4h  __attribute__((ext_vector_type(4)));
typedef float    v8f  __attribute__((ext_vector_type(8)));

union FragAB { v16h v; v8h h8[2]; uint4 q[2]; };

__device__ __forceinline__ float siluf(float x) {
    return x / (1.0f + __expf(-x));
}

// --- async copy of one 32KB packed-weight layer into LDS -------------------
// Pure byte copy (weights already f16): GL2 -> LDS without touching VGPRs,
// tracked on ASYNCcnt; overlaps with X-tile conversion work.
__device__ __forceinline__ void asyncStageWeights(const uint4* __restrict__ gsrc,
                                                  uint4* lds, int tid) {
    unsigned ldsBase = (unsigned)(size_t)lds;      // low 32 bits = LDS address
#pragma unroll
    for (int k = 0; k < 8; ++k) {
        int i = tid + k * 256;                     // 2048 x 16B = 32KB
        unsigned long long ga = (unsigned long long)(size_t)(gsrc + i);
        unsigned la = ldsBase + i * 16;
        asm volatile("global_load_async_to_lds_b128 %0, %1, off"
                     :: "v"(la), "v"(ga) : "memory");
    }
}

__device__ __forceinline__ void asyncWait() {
    asm volatile("s_wait_asynccnt 0x0" ::: "memory");
}

// --- A fragment (16x32 f16) from LDS, row-major halves, stride ROWH --------
// lanes 0-15: M=lane, VGPR0-3 K=0..7, VGPR4-7 K=8..15
// lanes16-31: M=lane-16, VGPR0-3 K=16..23, VGPR4-7 K=24..31
__device__ __forceinline__ v16h loadA(const _Float16* Xh, int rowBase,
                                      int kb, int lane) {
    const int m  = rowBase + (lane & 15);
    const int hs = lane >> 4;
    FragAB f;
    const v8h* p = reinterpret_cast<const v8h*>(Xh + m * ROWH + kb * 32 + hs * 16);
    f.h8[0] = p[0];      // K = kb*32 + hs*16 + 0..7
    f.h8[1] = p[1];      // K = kb*32 + hs*16 + 8..15   (+8 halves)
    return f.v;
}

// --- B fragment (32x16 f16) from LDS-staged pre-swizzled weights -----------
// Each lane's 8 dwords are contiguous -> 2x ds_load_b128.
__device__ __forceinline__ v16h loadB(const uint4* PW, int nb, int kb, int lane) {
    FragAB f;
    const v8h* p = reinterpret_cast<const v8h*>(PW + ((nb * 4 + kb) * 32 + lane) * 2);
    f.h8[0] = p[0];
    f.h8[1] = p[1];
    return f.v;
}

// ---------------------------------------------------------------------------
// Kernel 0: zero output + segment scratch
// ---------------------------------------------------------------------------
__global__ __launch_bounds__(256) void k_init(float* out, float* segSum,
                                              unsigned* segMax) {
    int i = blockIdx.x * 256 + threadIdx.x;
    const int GD = GG * DD;
    if (i < GD)                 out[i] = 0.0f;
    else if (i < GD + GG)       segSum[i - GD] = 0.0f;
    else if (i < GD + 2 * GG)   segMax[i - GD - GG] = 0u;  // encoded -inf
}

// ---------------------------------------------------------------------------
// Kernel 1: convert f32 weights -> f16 and swizzle into B-fragment layout
//   PW[((nb*4+kb)*32+lane)*8 + v] packs halves (k0, k0+1) of column n
//   k0 = kb*32 + (lane>>4)*16 + 2v,  n = nb*16 + (lane&15)
// ---------------------------------------------------------------------------
__global__ __launch_bounds__(256) void k_pack(const float* Wf1, const float* Wf2,
                                              const float* Ww1,
                                              unsigned* Pf1, unsigned* Pf2,
                                              unsigned* Pw1) {
    int id = blockIdx.x * 256 + threadIdx.x;      // 3 * 8192 total
    int wsel = id >> 13;
    int idx  = id & 8191;
    const float* W = (wsel == 0) ? Wf1 : (wsel == 1) ? Wf2 : Ww1;
    unsigned*  dst = (wsel == 0) ? Pf1 : (wsel == 1) ? Pf2 : Pw1;
    int v    = idx & 7;
    int lane = (idx >> 3) & 31;
    int kb   = (idx >> 8) & 3;
    int nb   = (idx >> 10) & 7;
    int k0 = kb * 32 + (lane >> 4) * 16 + 2 * v;
    int n  = nb * 16 + (lane & 15);
    union { _Float16 h[2]; unsigned u; } pk;
    pk.h[0] = (_Float16)W[(size_t)k0 * DD + n];
    pk.h[1] = (_Float16)W[(size_t)(k0 + 1) * DD + n];
    dst[idx] = pk.u;
}

// ---------------------------------------------------------------------------
// shared X-tile staging: 128 rows f32 -> f16 LDS (coalesced b128 loads)
// ---------------------------------------------------------------------------
__device__ __forceinline__ void stageX(const float* __restrict__ X, int base,
                                       _Float16* Xh, int tid) {
    const float4* src = reinterpret_cast<const float4*>(X + (size_t)base * DD);
    for (int i = tid; i < 128 * DD / 4; i += 256) {
        float4 f = src[i];
        int row  = i >> 5;
        int col4 = (i & 31) * 4;
        v4h h = { (_Float16)f.x, (_Float16)f.y, (_Float16)f.z, (_Float16)f.w };
        *reinterpret_cast<v4h*>(&Xh[row * ROWH + col4]) = h;
    }
}

// ---------------------------------------------------------------------------
// Kernel 2: per-node logit  w = silu(X@Ww1+bw1)@Ww2 + bw2, + segment atomicMax
//   1 workgroup = 128 nodes, 8 waves; each wave owns a 16-row strip.
// ---------------------------------------------------------------------------
__global__ __launch_bounds__(256) void k_logits(const float* __restrict__ X,
                                                const int* __restrict__ seg,
                                                const uint4* __restrict__ PWw1,
                                                const float* __restrict__ bw1,
                                                const float* __restrict__ ww2,
                                                const float* __restrict__ bw2,
                                                float* __restrict__ wOut,
                                                unsigned* __restrict__ segMax) {
    __shared__ _Float16 Xh[128 * ROWH];
    __shared__ uint4    PWlds[2048];              // 32KB packed Ww1
    const int base = blockIdx.x * 128;

    asyncStageWeights(PWw1, PWlds, threadIdx.x);  // ASYNCcnt copy, overlapped
    stageX(X, base, Xh, threadIdx.x);
    asyncWait();
    __syncthreads();

    const int wave = threadIdx.x >> 5;
    const int lane = threadIdx.x & 31;
    const int ln   = lane & 15;
    const int hs   = lane >> 4;
    const int rowBase = wave * 16;

    v16h Af[4];
#pragma unroll
    for (int kb = 0; kb < 4; ++kb) Af[kb] = loadA(Xh, rowBase, kb, lane);

    v8f acc[8];
    const v8f z8 = {0.f, 0.f, 0.f, 0.f, 0.f, 0.f, 0.f, 0.f};
#pragma unroll
    for (int nb = 0; nb < 8; ++nb) acc[nb] = z8;

#pragma unroll
    for (int kb = 0; kb < 4; ++kb) {
#pragma unroll
        for (int nb = 0; nb < 8; ++nb) {
            v16h b = loadB(PWlds, nb, kb, lane);
            acc[nb] = __builtin_amdgcn_wmma_f32_16x16x32_f16(
                false, Af[kb], false, b, (short)0, acc[nb], false, false);
        }
    }

    // epilogue: silu + dot with Ww2, butterfly-reduce across 16 columns
    float bw1v[8], ww2v[8];
#pragma unroll
    for (int nb = 0; nb < 8; ++nb) {
        bw1v[nb] = bw1[nb * 16 + ln];
        ww2v[nb] = ww2[nb * 16 + ln];
    }
    float pr[8];
#pragma unroll
    for (int r = 0; r < 8; ++r) {
        float s = 0.f;
#pragma unroll
        for (int nb = 0; nb < 8; ++nb)
            s += siluf(acc[nb][r] + bw1v[nb]) * ww2v[nb];
#pragma unroll
        for (int m = 1; m < 16; m <<= 1) s += __shfl_xor(s, m, 16);
        pr[r] = s;
    }
    if (ln == 0) {
        const float b2 = bw2[0];
#pragma unroll
        for (int r = 0; r < 8; ++r) {
            int row  = rowBase + hs * 8 + r;
            int node = base + row;
            float wv = pr[r] + b2;
            wOut[node] = wv;
            unsigned u = __float_as_uint(wv);
            unsigned enc = (u & 0x80000000u) ? ~u : (u | 0x80000000u);
            atomicMax(&segMax[seg[node]], enc);
        }
    }
}

// ---------------------------------------------------------------------------
// Kernel 3: w_exp = exp(w - segMax);  segSum += w_exp   (in-place on w)
// ---------------------------------------------------------------------------
__global__ __launch_bounds__(256) void k_expsum(float* __restrict__ w,
                                                const int* __restrict__ seg,
                                                const unsigned* __restrict__ segMax,
                                                float* __restrict__ segSum) {
    int i = blockIdx.x * 256 + threadIdx.x;
    if (i < NN) {
        int s = seg[i];
        unsigned k = segMax[s];
        unsigned b = (k & 0x80000000u) ? (k & 0x7FFFFFFFu) : ~k;
        float e = __expf(w[i] - __uint_as_float(b));
        w[i] = e;
        atomicAdd(&segSum[s], e);
    }
}

// ---------------------------------------------------------------------------
// Kernel 4: pf = silu(X@Wf1+bf1)@Wf2+bf2 recomputed, weighted segment pool.
// ---------------------------------------------------------------------------
__global__ __launch_bounds__(256) void k_pool(const float* __restrict__ X,
                                              const int* __restrict__ seg,
                                              const uint4* __restrict__ PWf1,
                                              const uint4* __restrict__ PWf2,
                                              const float* __restrict__ bf1,
                                              const float* __restrict__ bf2,
                                              const float* __restrict__ wExp,
                                              const float* __restrict__ segSum,
                                              float* __restrict__ out) {
    __shared__ _Float16 Xh[128 * ROWH];
    __shared__ uint4    PW1lds[2048];             // 32KB packed Wf1
    __shared__ uint4    PW2lds[2048];             // 32KB packed Wf2
    __shared__ float    swS[128];
    __shared__ int      segS[128];
    const int base = blockIdx.x * 128;

    asyncStageWeights(PWf1, PW1lds, threadIdx.x); // ASYNCcnt copies,
    asyncStageWeights(PWf2, PW2lds, threadIdx.x); // overlapped with cvt work
    stageX(X, base, Xh, threadIdx.x);
    if (threadIdx.x < 128) {
        int node = base + threadIdx.x;
        int s = seg[node];
        segS[threadIdx.x] = s;
        swS[threadIdx.x]  = wExp[node] / (segSum[s] + 1e-16f);
    }
    asyncWait();
    __syncthreads();

    const int wave = threadIdx.x >> 5;
    const int lane = threadIdx.x & 31;
    const int ln   = lane & 15;
    const int hs   = lane >> 4;
    const int rowBase = wave * 16;
    const v8f z8 = {0.f, 0.f, 0.f, 0.f, 0.f, 0.f, 0.f, 0.f};

    // ---- layer 1: h1 = silu(X @ Wf1 + bf1) ----
    v16h Af[4];
#pragma unroll
    for (int kb = 0; kb < 4; ++kb) Af[kb] = loadA(Xh, rowBase, kb, lane);

    v8f acc[8];
#pragma unroll
    for (int nb = 0; nb < 8; ++nb) acc[nb] = z8;
#pragma unroll
    for (int kb = 0; kb < 4; ++kb) {
#pragma unroll
        for (int nb = 0; nb < 8; ++nb) {
            v16h b = loadB(PW1lds, nb, kb, lane);
            acc[nb] = __builtin_amdgcn_wmma_f32_16x16x32_f16(
                false, Af[kb], false, b, (short)0, acc[nb], false, false);
        }
    }
    // silu + f16, overwrite our own 16 rows of Xh in place
    // (each wave reads/writes only rows [rowBase, rowBase+16) -> no barrier)
#pragma unroll
    for (int nb = 0; nb < 8; ++nb) {
        int n = nb * 16 + ln;
        float bias = bf1[n];
#pragma unroll
        for (int r = 0; r < 8; ++r) {
            int row = rowBase + hs * 8 + r;
            Xh[row * ROWH + n] = (_Float16)siluf(acc[nb][r] + bias);
        }
    }

    // ---- layer 2: pf = h1 @ Wf2 + bf2 ----
#pragma unroll
    for (int kb = 0; kb < 4; ++kb) Af[kb] = loadA(Xh, rowBase, kb, lane);
#pragma unroll
    for (int nb = 0; nb < 8; ++nb) acc[nb] = z8;
#pragma unroll
    for (int kb = 0; kb < 4; ++kb) {
#pragma unroll
        for (int nb = 0; nb < 8; ++nb) {
            v16h b = loadB(PW2lds, nb, kb, lane);
            acc[nb] = __builtin_amdgcn_wmma_f32_16x16x32_f16(
                false, Af[kb], false, b, (short)0, acc[nb], false, false);
        }
    }

    // ---- weighted segment sum, run-compressed along sorted rows ----
#pragma unroll
    for (int nb = 0; nb < 8; ++nb) {
        int n = nb * 16 + ln;
        float bias = bf2[n];
        int   curSeg = segS[rowBase + hs * 8];
        float run = 0.f;
#pragma unroll
        for (int r = 0; r < 8; ++r) {
            int row = rowBase + hs * 8 + r;
            int s   = segS[row];
            float v = (acc[nb][r] + bias) * swS[row];
            if (s != curSeg) {
                atomicAdd(&out[(size_t)curSeg * DD + n], run);
                run = 0.f;
                curSeg = s;
            }
            run += v;
        }
        atomicAdd(&out[(size_t)curSeg * DD + n], run);
    }
}

// ---------------------------------------------------------------------------
// Host launcher
// ---------------------------------------------------------------------------
extern "C" void kernel_launch(void* const* d_in, const int* in_sizes, int n_in,
                              void* d_out, int out_size, void* d_ws, size_t ws_size,
                              hipStream_t stream) {
    const float* X   = (const float*)d_in[0];
    const int*   seg = (const int*)d_in[1];
    const float* Wf1 = (const float*)d_in[2];
    const float* bf1 = (const float*)d_in[3];
    const float* Wf2 = (const float*)d_in[4];
    const float* bf2 = (const float*)d_in[5];
    const float* Ww1 = (const float*)d_in[6];
    const float* bw1 = (const float*)d_in[7];
    const float* Ww2 = (const float*)d_in[8];
    const float* bw2 = (const float*)d_in[9];
    float* out = (float*)d_out;

    // workspace layout
    char* w8 = (char*)d_ws;
    unsigned* Pw1 = (unsigned*)(w8 + 0);        // 32 KB packed Ww1 (f16)
    unsigned* Pf1 = (unsigned*)(w8 + 32768);    // 32 KB packed Wf1
    unsigned* Pf2 = (unsigned*)(w8 + 65536);    // 32 KB packed Wf2
    float*   wArr = (float*)(w8 + 98304);       // N f32 logits -> w_exp
    unsigned* sMx = (unsigned*)(w8 + 98304 + (size_t)NN * 4);
    float*   sSum = (float*)(w8 + 98304 + (size_t)NN * 4 + GG * 4);

    const int initElems = GG * DD + 2 * GG;
    k_init  <<<(initElems + 255) / 256, 256, 0, stream>>>(out, sSum, sMx);
    k_pack  <<<(3 * 8192) / 256, 256, 0, stream>>>(Wf1, Wf2, Ww1, Pf1, Pf2, Pw1);
    k_logits<<<TILES, 256, 0, stream>>>(X, seg, (const uint4*)Pw1,
                                        bw1, Ww2, bw2, wArr, sMx);
    k_expsum<<<NN / 256, 256, 0, stream>>>(wArr, seg, sMx, sSum);
    k_pool  <<<TILES, 256, 0, stream>>>(X, seg, (const uint4*)Pf1,
                                        (const uint4*)Pf2, bf1, bf2,
                                        wArr, sSum, out);
}
